// MultiHeadLatentAttention_33122787787445
// MI455X (gfx1250) — matrix operand order repaired and checked by reading, not compiled, not run on hardware
//
#include <hip/hip_runtime.h>

// ---- problem constants (match reference) ----
#define DM    1024   // d_model
#define DLAT  256    // d_latent
#define NH    16     // heads
#define HDIM  64     // head dim
#define BATCH 4
#define SEQ   2048

typedef __attribute__((ext_vector_type(16))) __bf16 v16bf;
typedef __attribute__((ext_vector_type(8)))  float  v8f;

union Frag16 { v16bf v; uint4 u[2]; };

__device__ __forceinline__ unsigned short f2bf(float x) {
  unsigned int u = __float_as_uint(x);
  u += 0x7FFFu + ((u >> 16) & 1u);          // round-to-nearest-even
  return (unsigned short)(u >> 16);
}

// A-operand fragment (16x32 bf16): lane holds row m=lane%16; h=lane/16 selects
// K = {8h..8h+7, 16+8h..16+8h+7}. Caller passes row_ptr + 8*h.
__device__ __forceinline__ v16bf ldA_frag(const unsigned short* p) {
  Frag16 f;
  f.u[0] = *(const uint4*)(p);
  f.u[1] = *(const uint4*)(p + 16);
  return f.v;
}
// B-operand fragment (32x16 bf16): lane holds col n=lane%16; h=lane/16 selects
// K = {8h..8h+7, 16+8h..16+8h+7}. Caller passes row_ptr + 8*h.
__device__ __forceinline__ v16bf ldB_frag(const unsigned short* p) {
  Frag16 f;
  f.u[0] = *(const uint4*)(p);
  f.u[1] = *(const uint4*)(p + 16);
  return f.v;
}

__device__ __forceinline__ v8f wmma_bf16(v16bf a, v16bf b, v8f c) {
  return __builtin_amdgcn_wmma_f32_16x16x32_bf16(
      /*neg_a=*/false, a, /*neg_b=*/false, b,
      /*c_mod=*/(short)0, c, /*reuse_a=*/false, /*reuse_b=*/false);
}

// ---- CDNA5 async global->LDS copy (ASYNCcnt path), per-lane 16 bytes ----
// LDS byte offset = low 32 bits of generic shared pointer (flat LDS aperture).
__device__ __forceinline__ void async_g2l_b128(const unsigned short* gp,
                                               unsigned short* lp) {
  unsigned long long ga = (unsigned long long)gp;
  unsigned la = (unsigned)(unsigned long long)lp;
  asm volatile("global_load_async_to_lds_b128 %0, %1, off"
               :: "v"(la), "v"(ga) : "memory");
}
__device__ __forceinline__ void wait_async0() {
  asm volatile("s_wait_asynccnt 0" ::: "memory");
}

// ---- 16-lane-group butterfly reductions via ds_swizzle (no index VGPR) ----
template <int MASK>
__device__ __forceinline__ float swz_xor(float v) {
#if __has_builtin(__builtin_amdgcn_ds_swizzle)
  return __int_as_float(
      __builtin_amdgcn_ds_swizzle(__float_as_int(v), 0x1F | (MASK << 10)));
#else
  return __shfl_xor(v, MASK, 32);
#endif
}
__device__ __forceinline__ float rmax16(float v) {
  v = fmaxf(v, swz_xor<1>(v));
  v = fmaxf(v, swz_xor<2>(v));
  v = fmaxf(v, swz_xor<4>(v));
  v = fmaxf(v, swz_xor<8>(v));
  return v;
}
__device__ __forceinline__ float rsum16(float v) {
  v += swz_xor<1>(v);
  v += swz_xor<2>(v);
  v += swz_xor<4>(v);
  v += swz_xor<8>(v);
  return v;
}

// ---------------- conversion kernels ----------------
__global__ void cvt_f32_bf16_kernel(const float* __restrict__ in,
                                    unsigned short* __restrict__ out, int n) {
  int i = blockIdx.x * blockDim.x + threadIdx.x;
  if (i < n) out[i] = f2bf(in[i]);
}

// W[K][N] fp32 -> Wt[N][K] bf16 (coalesced writes)
__global__ void transpose_cvt_kernel(const float* __restrict__ in,
                                     unsigned short* __restrict__ out,
                                     int K, int N) {
  int i = blockIdx.x * blockDim.x + threadIdx.x;
  if (i >= N * K) return;
  int n = i / K, k = i % K;
  out[i] = f2bf(in[(long)k * N + n]);
}

// ---------------- tiled bf16 GEMM: C = A[M,K] * Bt[N,K]^T + bias ----------------
// 256 threads = 8 waves; block tile 128x128x32; wave tile 32x64 (2x4 WMMA tiles).
// Tiles staged with async global->LDS copies (no VGPR round trip).
template <bool BF16_OUT>
__global__ __launch_bounds__(256)
void gemm_bf16_kernel(const unsigned short* __restrict__ A,
                      const unsigned short* __restrict__ Bt,
                      const float* __restrict__ bias,
                      void* __restrict__ Cout,
                      int M, int N, int K) {
  __shared__ __align__(16) unsigned short As[128 * 32];
  __shared__ __align__(16) unsigned short Bs[128 * 32];
  const int tid  = threadIdx.x;
  const int lane = tid & 31;
  const int wave = tid >> 5;
  const int wm = wave >> 1;      // 0..3 -> 32-row slab
  const int wn = wave & 1;       // 0..1 -> 64-col slab
  const int lr = lane & 15;
  const int lh = lane >> 4;
  const long bm = (long)blockIdx.y * 128;
  const long bn = (long)blockIdx.x * 128;

  v8f acc[2][4];
#pragma unroll
  for (int i = 0; i < 2; ++i)
#pragma unroll
    for (int j = 0; j < 4; ++j)
#pragma unroll
      for (int r = 0; r < 8; ++r) acc[i][j][r] = 0.0f;

  for (int k0 = 0; k0 < K; k0 += 32) {
#pragma unroll
    for (int c = 0; c < 2; ++c) {       // 512 16B chunks / 256 threads, x2 tiles
      int idx = tid + c * 256;
      int row = idx >> 2, seg = (idx & 3) * 8;
      async_g2l_b128(&A[(bm + row) * (long)K + k0 + seg], &As[row * 32 + seg]);
      async_g2l_b128(&Bt[(bn + row) * (long)K + k0 + seg], &Bs[row * 32 + seg]);
    }
    wait_async0();
    __syncthreads();
    v16bf af[2], bfv[4];
#pragma unroll
    for (int i = 0; i < 2; ++i)
      af[i] = ldA_frag(&As[(wm * 32 + i * 16 + lr) * 32 + 8 * lh]);
#pragma unroll
    for (int j = 0; j < 4; ++j)
      bfv[j] = ldB_frag(&Bs[(wn * 64 + j * 16 + lr) * 32 + 8 * lh]);
#pragma unroll
    for (int i = 0; i < 2; ++i)
#pragma unroll
      for (int j = 0; j < 4; ++j)
        acc[i][j] = wmma_bf16(af[i], bfv[j], acc[i][j]);
    __syncthreads();
  }

#pragma unroll
  for (int i = 0; i < 2; ++i) {
#pragma unroll
    for (int j = 0; j < 4; ++j) {
      long n  = bn + wn * 64 + j * 16 + lr;
      float b = bias[n];
      long m0 = bm + wm * 32 + i * 16 + 8 * lh;   // lanes 16-31 hold rows +8
#pragma unroll
      for (int r = 0; r < 8; ++r) {
        float v  = acc[i][j][r] + b;
        long off = (m0 + r) * (long)N + n;
        if (BF16_OUT) ((unsigned short*)Cout)[off] = f2bf(v);
        else          ((float*)Cout)[off]          = v;
      }
    }
  }
}

// ---------------- flash attention (online softmax, all-WMMA) ----------------
// Grid: (SEQ/64, NH, BATCH), 128 threads = 4 waves; wave w handles 16 q-rows.
// 64-key blocks: 8 score WMMAs + 8 PV WMMAs per block; one reduction tree
// per row per block (amortized over 4 score tiles).
__global__ __launch_bounds__(128)
void mla_attn_kernel(const unsigned short* __restrict__ Qb,
                     const unsigned short* __restrict__ Kb,
                     const unsigned short* __restrict__ Vb,
                     unsigned short* __restrict__ Ob) {
  __shared__ __align__(16) unsigned short Kt[64 * 64];      // [key][d]
  __shared__ __align__(16) unsigned short Vt[64 * 64];      // [d][key] (transposed)
  __shared__ __align__(16) unsigned short Pst[4][16 * 64];  // per-wave P staging
  const int tid  = threadIdx.x;
  const int lane = tid & 31;
  const int wave = tid >> 5;
  const int lr = lane & 15;
  const int lh = lane >> 4;
  const long rowbase = (long)blockIdx.z * SEQ;
  const long colbase = (long)blockIdx.y * HDIM;
  const int  qrow    = blockIdx.x * 64 + wave * 16;

  // Q fragments resident in registers: [16 q-rows x 64 d] = two 16x32 A-frags
  v16bf qf[2];
#pragma unroll
  for (int t = 0; t < 2; ++t)
    qf[t] = ldA_frag(&Qb[(rowbase + qrow + lr) * DM + colbase + t * 32 + 8 * lh]);

  v8f o[4];
  float Mst[8], Lst[8];
#pragma unroll
  for (int t = 0; t < 4; ++t)
#pragma unroll
    for (int r = 0; r < 8; ++r) o[t][r] = 0.0f;
#pragma unroll
  for (int r = 0; r < 8; ++r) { Mst[r] = -1e30f; Lst[r] = 0.0f; }

  unsigned short* P = &Pst[wave][0];
  // fold 1/sqrt(64) and log2(e) into one scale; stats kept in log2 domain
  const float SCL = 0.125f * 1.4426950408889634f;

  for (int kb = 0; kb < SEQ; kb += 64) {
    // async K tile load [64 key][64 d] (ASYNCcnt path)
#pragma unroll
    for (int c = 0; c < 4; ++c) {
      int idx = tid + c * 128;
      int row = idx >> 3, col = (idx & 7) * 8;
      async_g2l_b128(&Kb[(rowbase + kb + row) * DM + colbase + col],
                     &Kt[row * 64 + col]);
    }
    // V tile transposed [64 d][64 key]: vector global loads + strided LDS stores
#pragma unroll
    for (int c = 0; c < 4; ++c) {
      int idx = tid + c * 128;
      int key = idx >> 3, dseg = (idx & 7) * 8;
      union { uint4 q; unsigned short s[8]; } u;
      u.q = *(const uint4*)&Vb[(rowbase + kb + key) * DM + colbase + dseg];
#pragma unroll
      for (int j = 0; j < 8; ++j) Vt[(dseg + j) * 64 + key] = u.s[j];
    }
    wait_async0();
    __syncthreads();

    // scores: S[16q x 64key] = Q (16x64) * K^T -> four 16x16 C tiles
    v8f s[4];
#pragma unroll
    for (int kt = 0; kt < 4; ++kt)
#pragma unroll
      for (int r = 0; r < 8; ++r) s[kt][r] = 0.0f;
#pragma unroll
    for (int dh = 0; dh < 2; ++dh)
#pragma unroll
      for (int kt = 0; kt < 4; ++kt) {
        v16bf kf = ldB_frag(&Kt[(kt * 16 + lr) * 64 + dh * 32 + 8 * lh]);
        s[kt] = wmma_bf16(qf[dh], kf, s[kt]);
      }

    // online softmax; register r holds row r (lanes 0-15) / row r+8 (lanes 16-31)
    float alpha[8];
#pragma unroll
    for (int r = 0; r < 8; ++r) {
      float l0 = s[0][r] * SCL, l1 = s[1][r] * SCL;
      float l2 = s[2][r] * SCL, l3 = s[3][r] * SCL;
      float mx = rmax16(fmaxf(fmaxf(l0, l1), fmaxf(l2, l3)));
      float Mn = fmaxf(Mst[r], mx);
      float al = exp2f(Mst[r] - Mn);
      float p0 = exp2f(l0 - Mn), p1 = exp2f(l1 - Mn);
      float p2 = exp2f(l2 - Mn), p3 = exp2f(l3 - Mn);
      float sum = rsum16((p0 + p1) + (p2 + p3));
      Lst[r] = Lst[r] * al + sum;
      Mst[r] = Mn;
      alpha[r] = al;
      int m = r + 8 * lh;
      P[m * 64 +      lr] = f2bf(p0);
      P[m * 64 + 16 + lr] = f2bf(p1);
      P[m * 64 + 32 + lr] = f2bf(p2);
      P[m * 64 + 48 + lr] = f2bf(p3);
    }
#pragma unroll
    for (int t = 0; t < 4; ++t)
#pragma unroll
      for (int r = 0; r < 8; ++r) o[t][r] *= alpha[r];

    // O += P (16x64) * V (64x64): P as two A-frags, V rows in B-layout
    v16bf pf[2];
#pragma unroll
    for (int kh = 0; kh < 2; ++kh)
      pf[kh] = ldA_frag(&P[lr * 64 + kh * 32 + 8 * lh]);
#pragma unroll
    for (int t = 0; t < 4; ++t)
#pragma unroll
      for (int kh = 0; kh < 2; ++kh) {
        v16bf vf = ldB_frag(&Vt[(t * 16 + lr) * 64 + kh * 32 + 8 * lh]);
        o[t] = wmma_bf16(pf[kh], vf, o[t]);
      }
    __syncthreads();
  }

  // normalize and store bf16 O in [b, s, h*64+d] layout
#pragma unroll
  for (int t = 0; t < 4; ++t) {
#pragma unroll
    for (int r = 0; r < 8; ++r) {
      float inv = 1.0f / Lst[r];
      long  m   = rowbase + qrow + r + 8 * lh;
      Ob[m * DM + colbase + t * 16 + lr] = f2bf(o[t][r] * inv);
    }
  }
}

// ---------------- host orchestration ----------------
extern "C" void kernel_launch(void* const* d_in, const int* in_sizes, int n_in,
                              void* d_out, int out_size, void* d_ws, size_t ws_size,
                              hipStream_t stream) {
  (void)in_sizes; (void)n_in; (void)out_size; (void)ws_size;
  const float* x  = (const float*)d_in[0];
  const float* Wc = (const float*)d_in[1];
  const float* bc = (const float*)d_in[2];
  const float* Wk = (const float*)d_in[3];
  const float* bk = (const float*)d_in[4];
  const float* Wv = (const float*)d_in[5];
  const float* bv = (const float*)d_in[6];
  const float* Wq = (const float*)d_in[7];
  const float* bq = (const float*)d_in[8];
  const float* Wo = (const float*)d_in[9];
  const float* bo = (const float*)d_in[10];
  float* out = (float*)d_out;

  const long M = (long)BATCH * SEQ;   // 8192 rows
  char* ws = (char*)d_ws;
  size_t off = 0;
  auto alloc = [&](size_t bytes) {
    size_t o = off;
    off += (bytes + 255) & ~(size_t)255;
    return o;
  };
  unsigned short* xb  = (unsigned short*)(ws + alloc(M * DM   * 2));
  unsigned short* Qb  = (unsigned short*)(ws + alloc(M * DM   * 2));
  unsigned short* Lb  = (unsigned short*)(ws + alloc(M * DLAT * 2));
  unsigned short* Kb  = (unsigned short*)(ws + alloc(M * DM   * 2));
  unsigned short* Vb  = (unsigned short*)(ws + alloc(M * DM   * 2));
  unsigned short* Ob  = (unsigned short*)(ws + alloc(M * DM   * 2));
  unsigned short* Wqt = (unsigned short*)(ws + alloc((size_t)DM   * DM   * 2));
  unsigned short* Wct = (unsigned short*)(ws + alloc((size_t)DM   * DLAT * 2));
  unsigned short* Wkt = (unsigned short*)(ws + alloc((size_t)DLAT * DM   * 2));
  unsigned short* Wvt = (unsigned short*)(ws + alloc((size_t)DLAT * DM   * 2));
  unsigned short* Wot = (unsigned short*)(ws + alloc((size_t)DM   * DM   * 2));

  // 1) activations + weights -> bf16 (weights transposed to [N][K])
  {
    int n = (int)(M * DM);
    cvt_f32_bf16_kernel<<<(n + 255) / 256, 256, 0, stream>>>(x, xb, n);
  }
  transpose_cvt_kernel<<<(DM * DM   + 255) / 256, 256, 0, stream>>>(Wq, Wqt, DM,   DM);
  transpose_cvt_kernel<<<(DM * DLAT + 255) / 256, 256, 0, stream>>>(Wc, Wct, DM,   DLAT);
  transpose_cvt_kernel<<<(DLAT * DM + 255) / 256, 256, 0, stream>>>(Wk, Wkt, DLAT, DM);
  transpose_cvt_kernel<<<(DLAT * DM + 255) / 256, 256, 0, stream>>>(Wv, Wvt, DLAT, DM);
  transpose_cvt_kernel<<<(DM * DM   + 255) / 256, 256, 0, stream>>>(Wo, Wot, DM,   DM);

  // 2) projections (bf16 WMMA, fp32 accumulate, bf16 outputs)
  gemm_bf16_kernel<true><<<dim3(DM / 128,   M / 128), 256, 0, stream>>>(
      xb, Wqt, bq, Qb, (int)M, DM, DM);
  gemm_bf16_kernel<true><<<dim3(DLAT / 128, M / 128), 256, 0, stream>>>(
      xb, Wct, bc, Lb, (int)M, DLAT, DM);
  gemm_bf16_kernel<true><<<dim3(DM / 128,   M / 128), 256, 0, stream>>>(
      Lb, Wkt, bk, Kb, (int)M, DM, DLAT);
  gemm_bf16_kernel<true><<<dim3(DM / 128,   M / 128), 256, 0, stream>>>(
      Lb, Wvt, bv, Vb, (int)M, DM, DLAT);

  // 3) flash attention
  mla_attn_kernel<<<dim3(SEQ / 64, NH, BATCH), 128, 0, stream>>>(Qb, Kb, Vb, Ob);

  // 4) output projection (fp32 output to d_out)
  gemm_bf16_kernel<false><<<dim3(DM / 128,  M / 128), 256, 0, stream>>>(
      Ob, Wot, bo, out, (int)M, DM, DM);
}
